// MaSAd_3685081940151
// MI455X (gfx1250) — matrix operand order repaired and checked by reading, not compiled, not run on hardware
//
#include <hip/hip_runtime.h>
#include <hip/hip_bf16.h>

typedef __bf16 bf16;
typedef __attribute__((ext_vector_type(16))) __bf16 v16bf;
typedef __attribute__((ext_vector_type(8)))  __bf16 v8bf;
typedef __attribute__((ext_vector_type(8)))  float  v8f;
typedef __attribute__((ext_vector_type(4)))  int    v4i;
typedef __attribute__((ext_vector_type(4)))  unsigned int v4u;
typedef __attribute__((ext_vector_type(8)))  int    v8i;

#define NB   8
#define SH   64
#define SW   64
#define CC   256
#define NHD  8
#define KD   32
#define TT   (NB * SH * SW)   // 32768 tokens

// ---- CDNA5 async global->LDS + TDM paths (with safe fallbacks) ------------
#if defined(__has_builtin)
#if __has_builtin(__builtin_amdgcn_global_load_async_to_lds_b128)
#define HAVE_ASYNC_LDS 1
#endif
#if __has_builtin(__builtin_amdgcn_s_wait_asynccnt)
#define HAVE_WAIT_ASYNC 1
#endif
#if __has_builtin(__builtin_amdgcn_tensor_load_to_lds)
#define HAVE_TDM 1
#endif
#if __has_builtin(__builtin_amdgcn_s_wait_tensorcnt)
#define HAVE_WAIT_TENSOR 1
#endif
#endif

typedef __attribute__((address_space(1))) v4i*  gv4i_p;  // global int4*
typedef __attribute__((address_space(3))) v4i*  lv4i_p;  // LDS int4*
typedef __attribute__((address_space(3))) void* as3_vp;  // LDS void*

static __device__ __forceinline__ void g2l_b128(void* lds_dst, const void* gsrc) {
#ifdef HAVE_ASYNC_LDS
  __builtin_amdgcn_global_load_async_to_lds_b128(
      (gv4i_p)const_cast<void*>(gsrc), (lv4i_p)lds_dst, 0, 0);
#else
  *reinterpret_cast<int4*>(lds_dst) = *reinterpret_cast<const int4*>(gsrc);
#endif
}

static __device__ __forceinline__ void async_fence() {
#ifdef HAVE_WAIT_ASYNC
  __builtin_amdgcn_s_wait_asynccnt(0);
#elif defined(HAVE_ASYNC_LDS)
  asm volatile("s_wait_asynccnt 0" ::: "memory");
#endif
}

static __device__ __forceinline__ void tensor_fence() {
#ifdef HAVE_WAIT_TENSOR
  __builtin_amdgcn_s_wait_tensorcnt(0);
#else
  asm volatile("s_wait_tensorcnt 0" ::: "memory");
#endif
}

#ifdef HAVE_TDM
// One 2D TDM descriptor load: tile_w bf16 elements per row, tile_h rows,
// row_stride in elements. D# per CDNA5 ISA ch.8: group0 = {count/flags,
// lds_addr, global_addr lo, global_addr hi | type=2}; group1 packs data_size,
// tensor dims (== tile dims), tile dims and dim0 stride. This toolchain's
// builtin takes 6 args (g0, g1, g2, g3, g4, cpol).
static __device__ __forceinline__ void tdm_load_2d(void* lds_dst, const void* gsrc,
                                                   unsigned tile_w, unsigned tile_h,
                                                   unsigned row_stride) {
  const unsigned lds_off = (unsigned)(size_t)(as3_vp)lds_dst;
  const unsigned long long ga = (unsigned long long)(size_t)gsrc;
  v4u g0;
  g0[0] = 1u;                                            // count=1, user descriptor
  g0[1] = lds_off;                                       // lds_addr (bytes)
  g0[2] = (unsigned)ga;                                  // global_addr[31:0]
  g0[3] = (unsigned)((ga >> 32) & 0x01FFFFFFu) | 0x80000000u;  // addr[56:32] | type=2
  v8i g1;
  g1[0] = 0x00010000;                                    // data_size=1 (2B), mask=0
  g1[1] = (int)((tile_w & 0xFFFFu) << 16);               // tensor_dim0 lo16
  g1[2] = (int)((tile_w >> 16) | ((tile_h & 0xFFFFu) << 16));   // dim0 hi | dim1 lo
  g1[3] = (int)((tile_h >> 16) | (tile_w << 16));        // dim1 hi | tile_dim0
  g1[4] = (int)(tile_h & 0xFFFFu);                       // tile_dim1 | tile_dim2=0
  g1[5] = (int)row_stride;                               // tensor_dim0_stride lo32
  g1[6] = 0;                                             // stride hi | dim1_stride lo
  g1[7] = 0;
  v4i gz = {0, 0, 0, 0};
  v8i gz8 = {0, 0, 0, 0, 0, 0, 0, 0};
  __builtin_amdgcn_tensor_load_to_lds(g0, g1, gz, gz, gz8, 0);
}
#endif

static __device__ __forceinline__ v8f zero8() {
  v8f z = {0.f, 0.f, 0.f, 0.f, 0.f, 0.f, 0.f, 0.f};
  return z;
}

// A fragment (16x32 bf16, M x K) gathered from row-major bf16 LDS tile.
static __device__ __forceinline__ v16bf a_frag_lds(const bf16* base, int row0, int ld, int k0) {
  const int lane = threadIdx.x & 31;
  const int m   = lane & 15;
  const int hlf = lane >> 4;
  const bf16* p = base + (row0 + m) * ld + k0 + hlf * 8;
  union { v16bf v; v8bf h[2]; } u;
  u.h[0] = *reinterpret_cast<const v8bf*>(p);
  u.h[1] = *reinterpret_cast<const v8bf*>(p + 16);
  return u.v;
}

// A fragment gathered from a row-major f32 LDS tile, converted bf16 in-register.
static __device__ __forceinline__ v16bf a_frag_lds_f32(const float* base, int row0, int ld) {
  const int lane = threadIdx.x & 31;
  const int m   = lane & 15;
  const int hlf = lane >> 4;
  const float* p = base + (row0 + m) * ld + hlf * 8;
  union { v16bf v; bf16 e[16]; } u;
#pragma unroll
  for (int i = 0; i < 8; ++i) u.e[i] = (bf16)p[i];
#pragma unroll
  for (int i = 0; i < 8; ++i) u.e[8 + i] = (bf16)p[16 + i];
  return u.v;
}

// B fragment (32x16 bf16, K x N): lane l holds column N=l%16, K in table order (8h+s, 16+8h+s).
static __device__ __forceinline__ v16bf b_frag(const bf16* base, int k0, int ld, int n0) {
  const int lane = threadIdx.x & 31;
  const int nl = lane & 15, hlf = lane >> 4;
  const bf16* p = base + (size_t)(k0 + hlf * 8) * ld + n0 + nl;
  union { v16bf v; bf16 e[16]; } u;
#pragma unroll
  for (int i = 0; i < 8; ++i) u.e[i] = p[(size_t)i * ld];
#pragma unroll
  for (int i = 0; i < 8; ++i) u.e[8 + i] = p[(size_t)(16 + i) * ld];
  return u.v;
}

static __device__ __forceinline__ v8f wmma_bf16(v16bf a, v16bf b, v8f c) {
  return __builtin_amdgcn_wmma_f32_16x16x32_bf16(false, a, false, b, (short)0, c, false, false);
}

// ---------------------------------------------------------------------------
// Kernel 1: convert + transpose the four 256x256 weights to bf16.
// ---------------------------------------------------------------------------
__global__ void __launch_bounds__(256) k_wt(const float* __restrict__ Wq,
                                            const float* __restrict__ Wk,
                                            const float* __restrict__ Wv,
                                            const float* __restrict__ Wo,
                                            bf16* __restrict__ WqT, bf16* __restrict__ WkT,
                                            bf16* __restrict__ WvT, bf16* __restrict__ WoT) {
  const float* src;
  bf16* dst;
  switch (blockIdx.y) {
    case 0:  src = Wq; dst = WqT; break;
    case 1:  src = Wk; dst = WkT; break;
    case 2:  src = Wv; dst = WvT; break;
    default: src = Wo; dst = WoT; break;
  }
  const int cin  = blockIdx.x;
  const int cout = threadIdx.x;
  dst[cin * CC + cout] = (bf16)src[cout * CC + cin];
}

// ---------------------------------------------------------------------------
// Kernel 2: fused Q/K/V projection GEMM (async-staged fp32 A tile).
// ---------------------------------------------------------------------------
__global__ void __launch_bounds__(128) k_qkv(const float* __restrict__ x,
                                             const bf16* __restrict__ WqT,
                                             const bf16* __restrict__ WkT,
                                             const bf16* __restrict__ WvT,
                                             const float* __restrict__ bq,
                                             const float* __restrict__ bk,
                                             const float* __restrict__ bv,
                                             bf16* __restrict__ qb,
                                             bf16* __restrict__ kb,
                                             bf16* __restrict__ vb) {
  __shared__ __align__(16) float xs[64 * 32];   // fp32 x tile (8 KB)
  const int tid  = threadIdx.x;
  const int wid  = tid >> 5;
  const int lane = tid & 31;
  const int hlf  = lane >> 4;
  const int nl   = lane & 15;
  const int t0   = blockIdx.x * 64;
  const int col0 = blockIdx.y * 64 + wid * 16;

  v8f aq[4], ak[4], av[4];
#pragma unroll
  for (int i = 0; i < 4; ++i) { aq[i] = zero8(); ak[i] = zero8(); av[i] = zero8(); }

  for (int k0 = 0; k0 < CC; k0 += 32) {
    // 512 x 16B chunks, fixed trip count (no per-chunk bound checks)
#pragma unroll
    for (int it = 0; it < 4; ++it) {
      const int i = it * 128 + tid;
      const int r = i >> 3, c4 = (i & 7) * 4;
      g2l_b128(&xs[r * 32 + c4], &x[(size_t)(t0 + r) * CC + k0 + c4]);
    }
    async_fence();
    __syncthreads();

    const v16bf bqf = b_frag(WqT, k0, CC, col0);
    const v16bf bkf = b_frag(WkT, k0, CC, col0);
    const v16bf bvf = b_frag(WvT, k0, CC, col0);
    if (k0 + 32 < CC) {
      __builtin_prefetch(WqT + (size_t)(k0 + 32 + lane) * CC + col0, 0, 1);
      __builtin_prefetch(WkT + (size_t)(k0 + 32 + lane) * CC + col0, 0, 1);
      __builtin_prefetch(WvT + (size_t)(k0 + 32 + lane) * CC + col0, 0, 1);
    }
#pragma unroll
    for (int rt = 0; rt < 4; ++rt) {
      const v16bf a = a_frag_lds_f32(xs, rt * 16, 32);
      aq[rt] = wmma_bf16(a, bqf, aq[rt]);
      ak[rt] = wmma_bf16(a, bkf, ak[rt]);
      av[rt] = wmma_bf16(a, bvf, av[rt]);
    }
    __syncthreads();
  }

  const float scal = 0.17677669529663687f;  // 32^-0.5
  const int col = col0 + nl;
  const float bqv = bq[col], bkv = bk[col], bvv = bv[col];
#pragma unroll
  for (int rt = 0; rt < 4; ++rt) {
#pragma unroll
    for (int e = 0; e < 8; ++e) {
      const int row = t0 + rt * 16 + e + 8 * hlf;
      const size_t o = (size_t)row * CC + col;
      qb[o] = (bf16)(aq[rt][e] + bqv);
      kb[o] = (bf16)((ak[rt][e] + bkv) * scal);
      vb[o] = (bf16)(av[rt][e] + bvv);
    }
  }
}

// ---------------------------------------------------------------------------
// Kernel 3: LePE 5x5 depthwise conv (pad 2) over V, fp32 accumulate.
// ---------------------------------------------------------------------------
__global__ void __launch_bounds__(256) k_lepe(const bf16* __restrict__ vb,
                                              const float* __restrict__ Wl,
                                              const float* __restrict__ bl,
                                              float* __restrict__ lepe) {
  const int n  = blockIdx.x;   // token index
  const int c  = threadIdx.x;
  const int b  = n >> 12;
  const int hh = (n >> 6) & 63;
  const int ww = n & 63;
  float sum = bl[c];
#pragma unroll
  for (int ky = 0; ky < 5; ++ky) {
    const int y = hh + ky - 2;
    if ((unsigned)y >= 64u) continue;
#pragma unroll
    for (int kx = 0; kx < 5; ++kx) {
      const int xx = ww + kx - 2;
      if ((unsigned)xx >= 64u) continue;
      sum += (float)vb[((size_t)((b * 64 + y) * 64 + xx)) * CC + c] *
             Wl[(ky * 5 + kx) * CC + c];
    }
  }
  lepe[(size_t)n * CC + c] = sum;
}

// ---------------------------------------------------------------------------
// Kernels 4/5: one-axis attention. Q/V tiles staged by the Tensor Data Mover
// (one 2D descriptor each, issued by wave0/wave1, TENSORcnt-fenced); K tile
// transposed manually; scores via WMMA with mask as the C operand.
// ---------------------------------------------------------------------------
template <bool AXIS_H>
__global__ void __launch_bounds__(128) k_attn(const bf16* __restrict__ qb,
                                              const bf16* __restrict__ kb,
                                              const bf16* __restrict__ vsrc,
                                              const float* __restrict__ mask,  // (8,64,64)
                                              const float* __restrict__ lepe,  // AXIS_H only
                                              bf16* __restrict__ outb) {
  __shared__ __align__(32) bf16 qs[64 * 32];    // tokens x d
  __shared__ __align__(32) bf16 kts[32 * 64];   // d x tokens
  __shared__ __align__(32) bf16 vs[64 * 32];    // tokens x d
  __shared__ float ss[64 * 64];                 // fp32 scores
  __shared__ __align__(32) bf16 ps[64 * 64];    // bf16 probs

  const int tid  = threadIdx.x;
  const int wid  = tid >> 5;
  const int lane = tid & 31;
  const int hlf  = lane >> 4;
  const int nl   = lane & 15;

  const int head  = blockIdx.x & 7;
  const int bfi   = blockIdx.x >> 3;
  const int b     = bfi >> 6;
  const int fixed = bfi & 63;

  size_t base, stride;
  if (AXIS_H) { base = (size_t)(b * 4096 + fixed) * CC + head * KD;       stride = (size_t)64 * CC; }
  else        { base = (size_t)((b * 64 + fixed) * 64) * CC + head * KD;  stride = CC; }

#ifdef HAVE_TDM
  if (wid == 0) {
    tdm_load_2d(qs, qb + base, 32u, 64u, (unsigned)stride);    // Q tile DMA
  } else if (wid == 1) {
    tdm_load_2d(vs, vsrc + base, 32u, 64u, (unsigned)stride);  // V tile DMA
  }
#else
#pragma unroll
  for (int it = 0; it < 2; ++it) {
    const int i = it * 128 + tid;
    const int tok = i >> 2, d8 = (i & 3) * 8;
    const size_t g = base + (size_t)tok * stride + d8;
    g2l_b128(&qs[tok * 32 + d8], &qb[g]);
    g2l_b128(&vs[tok * 32 + d8], &vsrc[g]);
  }
#endif

  // K tile transposed (d-major) for the score B fragment
#pragma unroll
  for (int it = 0; it < 4; ++it) {
    const int i = it * 128 + tid;
    const int tok = i >> 3, d0 = (i & 7) * 4;
    const bf16* gp = kb + base + (size_t)tok * stride + d0;
    const bf16 e0 = gp[0], e1 = gp[1], e2 = gp[2], e3 = gp[3];
    kts[(d0 + 0) * 64 + tok] = e0;
    kts[(d0 + 1) * 64 + tok] = e1;
    kts[(d0 + 2) * 64 + tok] = e2;
    kts[(d0 + 3) * 64 + tok] = e3;
  }

#ifdef HAVE_TDM
  if (wid < 2) tensor_fence();
#else
  async_fence();
#endif
  __syncthreads();

  const int r0 = wid * 16;
  const float* mh = mask + head * 64 * 64;
#pragma unroll
  for (int ct = 0; ct < 4; ++ct) {
    const int n0 = ct * 16;
    v8f c;
#pragma unroll
    for (int e = 0; e < 8; ++e)
      c[e] = mh[(r0 + e + 8 * hlf) * 64 + n0 + nl];     // mask as WMMA C operand
    const v16bf a  = a_frag_lds(qs, r0, 32, 0);
    const v16bf bb = b_frag(kts, 0, 64, n0);
    c = wmma_bf16(a, bb, c);
#pragma unroll
    for (int e = 0; e < 8; ++e)
      ss[(r0 + e + 8 * hlf) * 64 + n0 + nl] = c[e];
  }
  __syncthreads();

  if (tid < 64) {                                       // one thread per row
    float* row = ss + tid * 64;
    float m = -3.4e38f;
    for (int j = 0; j < 64; ++j) m = fmaxf(m, row[j]);
    float s = 0.f;
    for (int j = 0; j < 64; ++j) { const float e = __expf(row[j] - m); row[j] = e; s += e; }
    const float inv = 1.f / s;
    bf16* prow = ps + tid * 64;
    for (int j = 0; j < 64; ++j) prow[j] = (bf16)(row[j] * inv);
  }
  __syncthreads();

#pragma unroll
  for (int nt = 0; nt < 2; ++nt) {
    const int n0 = nt * 16;
    v8f acc = zero8();
#pragma unroll
    for (int ks = 0; ks < 2; ++ks) {
      const v16bf a  = a_frag_lds(ps, r0, 64, ks * 32);
      const v16bf bb = b_frag(vs, ks * 32, 32, n0);
      acc = wmma_bf16(a, bb, acc);
    }
#pragma unroll
    for (int e = 0; e < 8; ++e) {
      const int tok = r0 + e + 8 * hlf;
      const int d   = n0 + nl;
      const size_t g = base + (size_t)tok * stride + d;
      float v = acc[e];
      if constexpr (AXIS_H) v += lepe[g];
      outb[g] = (bf16)v;
    }
  }
}

// ---------------------------------------------------------------------------
// Kernel 6: output projection GEMM, fp32 result + bias. A tile async-staged.
// ---------------------------------------------------------------------------
__global__ void __launch_bounds__(128) k_out(const bf16* __restrict__ x2,
                                             const bf16* __restrict__ WoT,
                                             const float* __restrict__ bo,
                                             float* __restrict__ out) {
  __shared__ __align__(32) bf16 xs[64 * 32];
  const int tid  = threadIdx.x;
  const int wid  = tid >> 5;
  const int lane = tid & 31;
  const int hlf  = lane >> 4;
  const int nl   = lane & 15;
  const int t0   = blockIdx.x * 64;
  const int col0 = blockIdx.y * 64 + wid * 16;

  v8f acc[4];
#pragma unroll
  for (int i = 0; i < 4; ++i) acc[i] = zero8();

  for (int k0 = 0; k0 < CC; k0 += 32) {
#pragma unroll
    for (int it = 0; it < 2; ++it) {                    // 256 x 16B chunks
      const int i = it * 128 + tid;
      const int r = i >> 2, c8 = (i & 3) * 8;
      g2l_b128(&xs[r * 32 + c8], &x2[(size_t)(t0 + r) * CC + k0 + c8]);
    }
    async_fence();
    __syncthreads();

    const v16bf bwf = b_frag(WoT, k0, CC, col0);
    if (k0 + 32 < CC)
      __builtin_prefetch(WoT + (size_t)(k0 + 32 + lane) * CC + col0, 0, 1);
#pragma unroll
    for (int rt = 0; rt < 4; ++rt) {
      const v16bf a = a_frag_lds(xs, rt * 16, 32, 0);
      acc[rt] = wmma_bf16(a, bwf, acc[rt]);
    }
    __syncthreads();
  }

  const int col = col0 + nl;
  const float bias = bo[col];
#pragma unroll
  for (int rt = 0; rt < 4; ++rt) {
#pragma unroll
    for (int e = 0; e < 8; ++e) {
      const int row = t0 + rt * 16 + e + 8 * hlf;
      out[(size_t)row * CC + col] = acc[rt][e] + bias;
    }
  }
}

// ---------------------------------------------------------------------------
extern "C" void kernel_launch(void* const* d_in, const int* in_sizes, int n_in,
                              void* d_out, int out_size, void* d_ws, size_t ws_size,
                              hipStream_t stream) {
  (void)in_sizes; (void)n_in; (void)out_size; (void)ws_size;

  const float* x     = (const float*)d_in[0];
  const float* maskh = (const float*)d_in[1];
  const float* maskw = (const float*)d_in[2];
  const float* Wq = (const float*)d_in[3];  const float* bq = (const float*)d_in[4];
  const float* Wk = (const float*)d_in[5];  const float* bk = (const float*)d_in[6];
  const float* Wv = (const float*)d_in[7];  const float* bv = (const float*)d_in[8];
  const float* Wl = (const float*)d_in[9];  const float* bl = (const float*)d_in[10];
  const float* Wo = (const float*)d_in[11]; const float* bo = (const float*)d_in[12];

  char* ws = (char*)d_ws;
  const size_t TC = (size_t)TT * CC;
  bf16* qb   = (bf16*)ws;  ws += TC * 2;
  bf16* kb   = (bf16*)ws;  ws += TC * 2;
  bf16* vb   = (bf16*)ws;  ws += TC * 2;
  bf16* v1b  = (bf16*)ws;  ws += TC * 2;
  bf16* x2b  = (bf16*)ws;  ws += TC * 2;
  float* lepe = (float*)ws; ws += TC * 4;
  bf16* WqT = (bf16*)ws;   ws += (size_t)CC * CC * 2;
  bf16* WkT = (bf16*)ws;   ws += (size_t)CC * CC * 2;
  bf16* WvT = (bf16*)ws;   ws += (size_t)CC * CC * 2;
  bf16* WoT = (bf16*)ws;   ws += (size_t)CC * CC * 2;

  k_wt<<<dim3(CC, 4), 256, 0, stream>>>(Wq, Wk, Wv, Wo, WqT, WkT, WvT, WoT);
  k_qkv<<<dim3(TT / 64, 4), 128, 0, stream>>>(x, WqT, WkT, WvT, bq, bk, bv, qb, kb, vb);
  k_lepe<<<dim3(TT), 256, 0, stream>>>(vb, Wl, bl, lepe);
  k_attn<false><<<dim3(NB * SH * NHD), 128, 0, stream>>>(qb, kb, vb, maskw, nullptr, v1b);
  k_attn<true><<<dim3(NB * SW * NHD), 128, 0, stream>>>(qb, kb, v1b, maskh, lepe, x2b);
  k_out<<<dim3(TT / 64, 4), 128, 0, stream>>>(x2b, WoT, bo, (float*)d_out);
}
